// CPRLinearMultiPrecision_64330020159891
// MI455X (gfx1250) — matrix operand order repaired and checked by reading, not compiled, not run on hardware
//
#include <hip/hip_runtime.h>
#include <hip/hip_bf16.h>

#define K_IN   4096
#define N_OUT  11008
#define N_HIGH 2752
#define N_LOW  8256
#define M_TOK  4096   // B*S = 2*2048

typedef __attribute__((ext_vector_type(16))) __bf16 v16bf;
typedef __attribute__((ext_vector_type(8)))  __bf16 v8bf;
typedef __attribute__((ext_vector_type(8)))  float  v8f;

// ---------------------------------------------------------------------------
// CDNA5 async global->LDS staging (ASYNCcnt-tracked, bypasses VGPRs).
// Inline asm keeps this portable across ROCm7.2 / amdgpu-toolchain builtins.
// ---------------------------------------------------------------------------
__device__ __forceinline__ void async_ld_b128(unsigned lds_off, const void* g) {
  asm volatile("global_load_async_to_lds_b128 %0, %1, off"
               :: "v"(lds_off), "v"((unsigned long long)(uintptr_t)g)
               : "memory");
}
__device__ __forceinline__ void wait_async0() {
  asm volatile("s_wait_asynccnt 0x0" ::: "memory");
}

// ---------------------------------------------------------------------------
// Kernel 0: forward permutation from inverse permutation: perm[col_inv[j]] = j
// ---------------------------------------------------------------------------
__global__ void build_perm_kernel(const long long* __restrict__ col_inv,
                                  int* __restrict__ perm) {
  int j = blockIdx.x * blockDim.x + threadIdx.x;
  if (j < N_OUT) perm[(int)col_inv[j]] = j;
}

// ---------------------------------------------------------------------------
// Kernel 1: x fp32 -> bf16 (row-major M x K)
// ---------------------------------------------------------------------------
__global__ void x_to_bf16_kernel(const float* __restrict__ x,
                                 __bf16* __restrict__ xb) {
  size_t base = ((size_t)blockIdx.x * blockDim.x + threadIdx.x) * 8;
  v8bf v;
#pragma unroll
  for (int i = 0; i < 8; ++i) v[i] = (__bf16)x[base + i];
  *(v8bf*)(xb + base) = v;
}

// ---------------------------------------------------------------------------
// Kernel 2: dequantize both slabs, apply column permutation, write weight
// K-major: Wt[j][k] (bf16).  Coalesced code reads along concat columns,
// transpose through LDS, contiguous 128B-run writes along K per final column.
// ---------------------------------------------------------------------------
__global__ __launch_bounds__(256) void dequant_permuted_kernel(
    const int*   __restrict__ hw,   // (K, N_HIGH) int8 codes in int32
    const float* __restrict__ hs,   // (K/128, N_HIGH)
    const int*   __restrict__ lw,   // (K, N_LOW) 4-bit codes in int32
    const float* __restrict__ ls1,  // (K,)
    const float* __restrict__ ls2,  // (N_LOW,)
    const float* __restrict__ lz,   // (K,)
    const int*   __restrict__ perm, // concat col -> final col
    __bf16*      __restrict__ Wt)   // (N_OUT, K) bf16, K-major
{
  __shared__ __bf16 tile[64 * 33];
  __shared__ int jmap[32];
  const int tid = threadIdx.x;
  const int k0 = blockIdx.x * 64;
  const int c0 = blockIdx.y * 32;

  if (tid < 32) jmap[tid] = perm[c0 + tid];

  const int cl = tid & 31;
  const int kb = tid >> 5;
  const int c  = c0 + cl;
#pragma unroll
  for (int i = 0; i < 8; ++i) {
    const int kl = kb + i * 8;
    const int k  = k0 + kl;
    float v;
    if (c < N_HIGH) {
      v = (float)hw[(size_t)k * N_HIGH + c] * hs[(size_t)(k >> 7) * N_HIGH + c];
    } else {
      const int c2 = c - N_HIGH;
      v = ((float)lw[(size_t)k * N_LOW + c2] - lz[k]) * ls1[k] * ls2[c2];
    }
    tile[kl * 33 + cl] = (__bf16)v;
  }
  __syncthreads();

  const int cc   = tid >> 3;
  const int kblk = tid & 7;
  const int j = jmap[cc];
  v8bf o;
#pragma unroll
  for (int i = 0; i < 8; ++i) o[i] = tile[(kblk * 8 + i) * 33 + cc];
  *(v8bf*)(Wt + (size_t)j * K_IN + k0 + kblk * 8) = o;
}

// ---------------------------------------------------------------------------
// Kernel 3: bf16 WMMA GEMM with async double-buffered LDS staging.
// Block tile 128(M) x 256(N), K-step 32.  8 waves in 2(M) x 4(N); each wave
// owns a 64x64 macro-tile = 4x4 v_wmma_f32_16x16x32_bf16 accumulators.
// ---------------------------------------------------------------------------
#define BM 128
#define BN 256
#define BK 32
#define LDT 40  // padded LDS row stride (bf16 elems); 80B, 16B-aligned

__global__ __launch_bounds__(256) void gemm_bf16_wmma_kernel(
    const __bf16* __restrict__ Xb,   // (M, K)
    const __bf16* __restrict__ Wt,   // (N, K)
    const float*  __restrict__ bias, // (N,)
    float*        __restrict__ out)  // (M, N)
{
  __shared__ __align__(16) __bf16 As[2 * BM * LDT];
  __shared__ __align__(16) __bf16 Bs[2 * BN * LDT];

  const int tid  = threadIdx.x;
  const int m0   = blockIdx.y * BM;
  const int n0   = blockIdx.x * BN;
  const int wave = tid >> 5;
  const int lane = tid & 31;
  const int wm   = wave & 1;   // 2 wave-rows of 64
  const int wn   = wave >> 1;  // 4 wave-cols of 64
  const int lrow  = lane & 15;
  const int lhalf = lane >> 4;

  const unsigned asBase = (unsigned)(uintptr_t)&As[0];
  const unsigned bsBase = (unsigned)(uintptr_t)&Bs[0];

  // per-thread staging coordinates: 16B chunks
  const int ar = tid >> 2;          // A row 0..63 (x2 passes -> 128)
  const int akc = (tid & 3) << 3;   // A k-chunk

  // stage one K-slab (A: 128x32, B: 256x32) into LDS buffer `buf`
  auto stage = [&](int buf, int kt) {
#pragma unroll
    for (int p = 0; p < 2; ++p) {
      const int r = ar + p * 64;
      async_ld_b128(asBase + (unsigned)(buf * BM * LDT + r * LDT + akc) * 2,
                    &Xb[(size_t)(m0 + r) * K_IN + kt + akc]);
    }
#pragma unroll
    for (int p = 0; p < 4; ++p) {
      const int li = tid + p * 256;
      const int r  = li >> 2;
      const int kc = (li & 3) << 3;
      async_ld_b128(bsBase + (unsigned)(buf * BN * LDT + r * LDT + kc) * 2,
                    &Wt[(size_t)(n0 + r) * K_IN + kt + kc]);
    }
  };

  v8f acc[4][4] = {};

  stage(0, 0);
  wait_async0();
  __syncthreads();

  int buf = 0;
  for (int kt = 0; kt < K_IN; kt += BK) {
    if (kt + BK < K_IN) stage(buf ^ 1, kt + BK);  // fill other buffer async

    const __bf16* Ab = &As[buf * BM * LDT];
    const __bf16* Bb = &Bs[buf * BN * LDT];

    // A fragments: lane half h holds K in {8h..8h+7, 16+8h..16+8h+7}
    v16bf a[4];
#pragma unroll
    for (int i = 0; i < 4; ++i) {
      const int row = wm * 64 + i * 16 + lrow;
      v8bf lo = *(const v8bf*)&Ab[row * LDT + lhalf * 8];
      v8bf hi = *(const v8bf*)&Ab[row * LDT + 16 + lhalf * 8];
      a[i] = __builtin_shufflevector(lo, hi, 0, 1, 2, 3, 4, 5, 6, 7,
                                     8, 9, 10, 11, 12, 13, 14, 15);
    }
    // B fragments: lane l holds col l%16, K in {8h..8h+7, 16+8h..16+8h+7}
#pragma unroll
    for (int j = 0; j < 4; ++j) {
      const int nr = wn * 64 + j * 16 + lrow;
      v8bf lo = *(const v8bf*)&Bb[nr * LDT + lhalf * 8];
      v8bf hi = *(const v8bf*)&Bb[nr * LDT + 16 + lhalf * 8];
      v16bf b = __builtin_shufflevector(lo, hi, 0, 1, 2, 3, 4, 5, 6, 7,
                                        8, 9, 10, 11, 12, 13, 14, 15);
#pragma unroll
      for (int i = 0; i < 4; ++i)
        acc[i][j] = __builtin_amdgcn_wmma_f32_16x16x32_bf16(
            false, a[i], false, b, (short)0, acc[i][j], false, false);
    }

    wait_async0();   // our async fills for the other buffer have landed
    __syncthreads(); // everyone's fills visible; safe to flip
    buf ^= 1;
  }

  // epilogue: C layout lane l%16 = n, VGPR r -> m = r + 8*(l/16)
#pragma unroll
  for (int j = 0; j < 4; ++j) {
    const int col = n0 + wn * 64 + j * 16 + lrow;
    const float bv = bias[col];
#pragma unroll
    for (int i = 0; i < 4; ++i) {
      const int mbase = m0 + wm * 64 + i * 16 + 8 * lhalf;
#pragma unroll
      for (int r = 0; r < 8; ++r)
        out[(size_t)(mbase + r) * N_OUT + col] = acc[i][j][r] + bv;
    }
  }
}

// ---------------------------------------------------------------------------
extern "C" void kernel_launch(void* const* d_in, const int* in_sizes, int n_in,
                              void* d_out, int out_size, void* d_ws, size_t ws_size,
                              hipStream_t stream) {
  const float*     x    = (const float*)d_in[0];
  const int*       hw   = (const int*)d_in[1];
  const float*     hs   = (const float*)d_in[2];
  const int*       lw   = (const int*)d_in[3];
  const float*     ls1  = (const float*)d_in[4];
  const float*     ls2  = (const float*)d_in[5];
  const float*     lz   = (const float*)d_in[6];
  const long long* cinv = (const long long*)d_in[7];
  const float*     bias = (const float*)d_in[8];
  float*           out  = (float*)d_out;

  // workspace layout: [Wt bf16 N*K | Xb bf16 M*K | perm int N]
  __bf16* Wt   = (__bf16*)d_ws;
  __bf16* Xb   = Wt + (size_t)N_OUT * K_IN;
  int*    perm = (int*)(Xb + (size_t)M_TOK * K_IN);

  build_perm_kernel<<<(N_OUT + 255) / 256, 256, 0, stream>>>(cinv, perm);

  x_to_bf16_kernel<<<(size_t)M_TOK * K_IN / 8 / 256, 256, 0, stream>>>(x, Xb);

  dim3 gdq(K_IN / 64, N_OUT / 32);
  dequant_permuted_kernel<<<gdq, 256, 0, stream>>>(hw, hs, lw, ls1, ls2, lz,
                                                   perm, Wt);

  dim3 gg(N_OUT / BN, M_TOK / BM);
  gemm_bf16_wmma_kernel<<<gg, 256, 0, stream>>>(Xb, Wt, bias, out);
}
